// XEGNNK_46454366274174
// MI455X (gfx1250) — matrix operand order repaired and checked by reading, not compiled, not run on hardware
//
#include <hip/hip_runtime.h>

// ---------------------------------------------------------------------------
// CDNA5 / gfx1250 EGNN layer. Edge MLPs on WMMA f16 (f32 accumulate).
// Weights pre-swizzled into WMMA B-fragment order (contiguous 32B per lane),
// activations staged in LDS in A-fragment order -> all fragment loads are
// b128, no per-element gather in the GEMM inner loops.
// ---------------------------------------------------------------------------

typedef _Float16 half_t;
typedef __attribute__((ext_vector_type(16))) _Float16 v16h;
typedef __attribute__((ext_vector_type(8)))  float    v8f;

#define KCH    128
#define DE     64
#define MIN_   260
#define MPAD   288   // 260 padded to multiple of 32
#define TILE   16
#define CLAMPV 10.0f

__device__ __forceinline__ float silu_f(float x) { return x / (1.0f + __expf(-x)); }

// ---------------------------------------------------------------------------
// Fragment-layout conventions (ISA 7.12.2, wave32):
//  A (16x32 f16), lane<16: M=lane,   halves 0..7 -> K 0..7,  8..15 -> K 16..23
//                 lane>=16: M=lane-16, K offset +8
//  => element (m, k) of chunk q lives at Afr[(q*32 + m + ((k&8)?16:0))*16
//                                            + (k&7) + ((k&16)?8:0)]
//  B (32x16 f16), lane<16: N=lane, halves 0..7 -> K 0..7, 8..15 -> K 16..23; lane>=16: K +8
//  => per (slab s, chunk q, lane, h): n = s*16+(lane&15),
//                                     k = q*32 + ((lane&16)?8:0) + (h&7) + ((h&8)?16:0)
//  C/D (16x16 f32): lane<16 -> N=lane, reg r -> M=r; lane>=16 -> M=r+8
// ---------------------------------------------------------------------------

// fragment-order GEMM: both A (LDS) and B (global, pre-swizzled) are
// contiguous 32-byte chunks per lane -> b128 loads.
__device__ __forceinline__ v8f gemm16_fr(const half_t* Afr, int Q,
                                         const half_t* Bsw, int slab,
                                         const float* bias, int lane) {
    v8f c;
    float bv = bias[slab * 16 + (lane & 15)];
#pragma unroll
    for (int r = 0; r < 8; ++r) c[r] = bv;
    for (int q = 0; q < Q; ++q) {
        v16h a = *(const v16h*)(Afr + (size_t)(q * 32 + lane) * 16);
        v16h b = *(const v16h*)(Bsw + (size_t)((slab * Q + q) * 32 + lane) * 16);
        c = __builtin_amdgcn_wmma_f32_16x16x32_f16(false, a, false, b, (short)0, c,
                                                   false, false);
    }
    return c;
}

// store D tile into the NEXT gemm's A-fragment layout (K dim = this N dim)
__device__ __forceinline__ void store_frag(half_t* Dst, int n0, v8f c, int lane, bool act) {
    const int n   = n0 + (lane & 15);
    const int q   = n >> 5;
    const int kin = n & 31;
    const int h   = (kin & 7) + ((kin & 16) ? 8 : 0);
    const int la  = (kin & 8) ? 16 : 0;
    const int mb  = (lane & 16) ? 8 : 0;
#pragma unroll
    for (int r = 0; r < 8; ++r) {
        float v = c[r];
        if (act) v = silu_f(v);
        Dst[(size_t)(q * 32 + (mb + r) + la) * 16 + h] = (half_t)v;
    }
}

__device__ __forceinline__ void store_f32_clip(float* D, int ld, int n0, v8f c, int lane) {
    const int n  = n0 + (lane & 15);
    const int mb = (lane & 16) ? 8 : 0;
#pragma unroll
    for (int r = 0; r < 8; ++r)
        D[(mb + r) * ld + n] = fminf(fmaxf(c[r], -CLAMPV), CLAMPV);
}

// ---------------------------------------------------------------------------
// Prep kernels
// ---------------------------------------------------------------------------
__global__ void zero_kernel(float* p, int n) {
    int i = blockIdx.x * blockDim.x + threadIdx.x;
    if (i < n) p[i] = 0.0f;
}

// weight [Ktot x Nout] (rows 'rows' valid, rest zero-pad) -> B-fragment order
__global__ void swz_kernel(const float* __restrict__ src, half_t* __restrict__ dst,
                           int rows, int Ktot, int Nout) {
    const int Q = Ktot / 32;
    const int total = Ktot * Nout;
    int i = blockIdx.x * blockDim.x + threadIdx.x;
    if (i >= total) return;
    int h     = i & 15;
    int lane  = (i >> 4) & 31;
    int chunk = i >> 9;            // = s*Q + q
    int q = chunk % Q, s = chunk / Q;
    int n = s * 16 + (lane & 15);
    int k = q * 32 + ((lane & 16) ? 8 : 0) + (h & 7) + ((h & 8) ? 16 : 0);
    dst[i] = (k < rows) ? (half_t)src[k * Nout + n] : (half_t)0.0f;
}

// per-graph sum of X and node counts
__global__ void sumx_kernel(const float* __restrict__ X, const int* __restrict__ batch,
                            float* __restrict__ sumX, float* __restrict__ cnt, int N) {
    int i = blockIdx.x * blockDim.x + threadIdx.x;
    if (i >= N * 384) return;
    int n = i / 384, c = i % 384;
    int b = batch[n];
    atomicAdd(&sumX[b * 384 + c], X[i]);
    if (c == 0) atomicAdd(&cnt[b], 1.0f);
}

// centered 3-vector norms, accumulated per graph per k
__global__ void norm_kernel(const float* __restrict__ X, const int* __restrict__ batch,
                            const float* __restrict__ sumX, const float* __restrict__ cnt,
                            float* __restrict__ nsum, int N) {
    int i = blockIdx.x * blockDim.x + threadIdx.x;
    if (i >= N * KCH) return;
    int n = i >> 7, k = i & 127;
    int b = batch[n];
    float ic = 1.0f / fmaxf(cnt[b], 1.0f);
    float x0 = X[n * 384 + k]       - sumX[b * 384 + k]       * ic;
    float x1 = X[n * 384 + 128 + k] - sumX[b * 384 + 128 + k] * ic;
    float x2 = X[n * 384 + 256 + k] - sumX[b * 384 + 256 + k] * ic;
    atomicAdd(&nsum[b * KCH + k], sqrtf(x0 * x0 + x1 * x1 + x2 * x2));
}

// E3Norm -> Xn (also seeds d_out), accumulate per-graph sum of Xn
__global__ void xn_kernel(const float* __restrict__ X, const int* __restrict__ batch,
                          const float* __restrict__ sumX, const float* __restrict__ cnt,
                          const float* __restrict__ nsum, const float* __restrict__ wxn,
                          float* __restrict__ Xn, float* __restrict__ sumXn,
                          float* __restrict__ out, int N) {
    int i = blockIdx.x * blockDim.x + threadIdx.x;
    if (i >= N * KCH) return;
    int n = i >> 7, k = i & 127;
    int b = batch[n];
    float ic = 1.0f / fmaxf(cnt[b], 1.0f);
    float mn = nsum[b * KCH + k] * ic;
    float sc = wxn[k] / (mn + 1e-5f);
#pragma unroll
    for (int d = 0; d < 3; ++d) {
        float xc = (X[n * 384 + d * 128 + k] - sumX[b * 384 + d * 128 + k] * ic) * sc;
        Xn[n * 384 + d * 128 + k]  = xc;
        out[n * 384 + d * 128 + k] = xc;
        atomicAdd(&sumXn[b * 384 + d * 128 + k], xc);
    }
}

__global__ void meanxn_kernel(float* __restrict__ sumXn, const float* __restrict__ cnt, int B) {
    int i = blockIdx.x * blockDim.x + threadIdx.x;
    if (i >= B * 384) return;
    sumXn[i] *= 1.0f / fmaxf(cnt[i / 384], 1.0f);
}

// dist_coord[n,i,d] = Xn[n,i,:] @ W_cp + b_cp
__global__ void dc_kernel(const float* __restrict__ Xn, const float* __restrict__ Wcp,
                          const float* __restrict__ bcp, float* __restrict__ dc, int N) {
    int i = blockIdx.x * blockDim.x + threadIdx.x;
    if (i >= N * 12) return;
    int n = i / 12, r = i % 12, ax = r >> 2, d = r & 3;
    float acc = bcp[d];
    for (int k = 0; k < KCH; ++k)
        acc += Xn[n * 384 + ax * 128 + k] * Wcp[k * 4 + d];
    dc[i] = acc;
}

// H projection (64x64) + SiLU + LayerNorm; one wave32 per node, 2 outputs/lane
__global__ __launch_bounds__(256)
void hproj_kernel(const float* __restrict__ H, const float* __restrict__ W,
                  const float* __restrict__ b, const float* __restrict__ gamma,
                  const float* __restrict__ beta, float* __restrict__ Hn, int N) {
    int wv = threadIdx.x >> 5, lane = threadIdx.x & 31;
    int node = blockIdx.x * 8 + wv;
    if (node >= N) return;
    int j0 = 2 * lane;
    float a0 = b[j0], a1 = b[j0 + 1];
    for (int k = 0; k < DE; ++k) {
        float h = H[node * DE + k];
        a0 += h * W[k * DE + j0];
        a1 += h * W[k * DE + j0 + 1];
    }
    a0 = silu_f(a0); a1 = silu_f(a1);
    float s = a0 + a1, s2 = a0 * a0 + a1 * a1;
    for (int m = 16; m >= 1; m >>= 1) {
        s  += __shfl_xor(s, m, 32);
        s2 += __shfl_xor(s2, m, 32);
    }
    float mu  = s * (1.0f / 64.0f);
    float var = s2 * (1.0f / 64.0f) - mu * mu;
    float inv = rsqrtf(var + 1e-5f);
    Hn[node * DE + j0]     = gamma[j0]     * (a0 - mu) * inv + beta[j0];
    Hn[node * DE + j0 + 1] = gamma[j0 + 1] * (a1 - mu) * inv + beta[j0 + 1];
}

// ---------------------------------------------------------------------------
// Main edge kernel: 128 threads = 4 wave32 per 16-edge tile, WMMA MLP chain.
// ---------------------------------------------------------------------------
#define QIN  (MPAD / 32)   // 9 k-chunks for the 288-wide input GEMM
#define QH   (DE / 32)     // 2 k-chunks for 64-wide GEMMs

__global__ __launch_bounds__(128)
void edge_kernel(const int* __restrict__ srcI, const int* __restrict__ tgtI,
                 const int* __restrict__ batch,
                 const float* __restrict__ Hn, const float* __restrict__ eattr,
                 const float* __restrict__ te, const float* __restrict__ dc,
                 const float* __restrict__ Xn, const float* __restrict__ meanXn,
                 const half_t* __restrict__ Wm1s, const float* __restrict__ bm1,
                 const half_t* __restrict__ Wm2s, const float* __restrict__ bm2,
                 const half_t* __restrict__ Wx1s, const float* __restrict__ bx1,
                 const half_t* __restrict__ Wx2s, const float* __restrict__ bx2,
                 const half_t* __restrict__ Wc1s, const float* __restrict__ bc1,
                 const half_t* __restrict__ Wc2s, const float* __restrict__ bc2,
                 float* __restrict__ out, int E) {
    __shared__ __align__(32) half_t AinF[QIN * 32 * 16];  // m_in, A-fragment order
    __shared__ __align__(32) half_t MijF[QH * 32 * 16];   // m_ij, A-fragment order
    __shared__ __align__(32) half_t TmpF[QH * 32 * 16];   // hidden, A-fragment order
    __shared__ float  CW [TILE * KCH];                    // coor_wij (clipped)
    __shared__ float  CWC[TILE * KCH];                    // coor_wij_cross (clipped)
    __shared__ int    sS[TILE], sT[TILE], sBs[TILE], sBt[TILE], sV[TILE];

    const int tid  = threadIdx.x;
    const int lane = tid & 31;
    const int wv   = tid >> 5;
    const int e0   = blockIdx.x * TILE;

    if (tid < TILE) {
        int ge = e0 + tid;
        int valid = (ge < E);
        int s = valid ? srcI[ge] : 0;
        int t = valid ? tgtI[ge] : 0;
        sS[tid] = s; sT[tid] = t;
        sBs[tid] = batch[s]; sBt[tid] = batch[t];
        sV[tid] = valid;
    }
    __syncthreads();

    // ---- stage m_in directly in A-fragment order --------------------------
    // idx -> (q, ln, h); edge m = ln&15, feature c = q*32 + kb + (h&7) + hi16
    for (int idx = tid; idx < QIN * 32 * 16; idx += 128) {
        int h  = idx & 15;
        int ln = (idx >> 4) & 31;
        int q  = idx >> 9;
        int e  = ln & 15;
        int c  = q * 32 + ((ln & 16) ? 8 : 0) + (h & 7) + ((h & 8) ? 16 : 0);
        float v;
        if (c < 64) {
            v = Hn[sT[e] * DE + c];
        } else if (c < 128) {
            v = Hn[sS[e] * DE + (c - 64)];
        } else if (c < 192) {
            int ge = sV[e] ? (e0 + e) : 0;
            v = eattr[(size_t)ge * DE + (c - 128)];
        } else if (c < 196) {
            int d = c - 192;
            float acc = 0.0f;
#pragma unroll
            for (int i = 0; i < 3; ++i) {
                float df = dc[sS[e] * 12 + i * 4 + d] - dc[sT[e] * 12 + i * 4 + d];
                acc += df * df;
            }
            v = acc;
        } else if (c < 260) {
            v = te[sBs[e] * DE + (c - 196)];
        } else {
            v = 0.0f;  // K padding 260..287
        }
        AinF[idx] = (half_t)v;
    }
    __syncthreads();

    // ---- phi_message: 288->64 (SiLU) -> 64->64 ----
    { v8f c = gemm16_fr(AinF, QIN, Wm1s, wv, bm1, lane);
      store_frag(TmpF, 16 * wv, c, lane, true); }
    __syncthreads();
    { v8f c = gemm16_fr(TmpF, QH, Wm2s, wv, bm2, lane);
      store_frag(MijF, 16 * wv, c, lane, false); }
    __syncthreads();

    // ---- phi_x: 64->64 (SiLU) -> 64->128, clip ----
    { v8f c = gemm16_fr(MijF, QH, Wx1s, wv, bx1, lane);
      store_frag(TmpF, 16 * wv, c, lane, true); }
    __syncthreads();
#pragma unroll
    for (int hf = 0; hf < 2; ++hf) {
        int slab = wv + 4 * hf;
        v8f c = gemm16_fr(TmpF, QH, Wx2s, slab, bx2, lane);
        store_f32_clip(CW, KCH, slab * 16, c, lane);
    }
    __syncthreads();

    // ---- phi_x_cross: 64->64 (SiLU) -> 64->128, clip ----
    { v8f c = gemm16_fr(MijF, QH, Wc1s, wv, bc1, lane);
      store_frag(TmpF, 16 * wv, c, lane, true); }
    __syncthreads();
#pragma unroll
    for (int hf = 0; hf < 2; ++hf) {
        int slab = wv + 4 * hf;
        v8f c = gemm16_fr(TmpF, QH, Wc2s, slab, bc2, lane);
        store_f32_clip(CWC, KCH, slab * 16, c, lane);
    }
    __syncthreads();

    // ---- geometric scatter: rel-coord branch + cross-product branch ----
    for (int it = 0; it < TILE * KCH / 128; ++it) {
        int idx = it * 128 + tid;
        int e = idx >> 7, k = idx & 127;
        if (!sV[e]) continue;
        int s = sS[e], t = sT[e], bs = sBs[e], bt = sBt[e];
        float xs0 = Xn[s * 384 + k],       xt0 = Xn[t * 384 + k];
        float xs1 = Xn[s * 384 + 128 + k], xt1 = Xn[t * 384 + 128 + k];
        float xs2 = Xn[s * 384 + 256 + k], xt2 = Xn[t * 384 + 256 + k];
        float r0 = xs0 - xt0, r1 = xs1 - xt1, r2 = xs2 - xt2;
        float rd = r0 * r0 + r1 * r1 + r2 * r2;
        float inv1 = 1.0f / (1.0f + sqrtf(rd + 1e-8f));
        float w1 = CW[e * KCH + k];

        float a0 = xs0 - meanXn[bs * 384 + k];
        float a1 = xs1 - meanXn[bs * 384 + 128 + k];
        float a2 = xs2 - meanXn[bs * 384 + 256 + k];
        float b0 = xt0 - meanXn[bt * 384 + k];
        float b1 = xt1 - meanXn[bt * 384 + 128 + k];
        float b2 = xt2 - meanXn[bt * 384 + 256 + k];
        float c0 = a1 * b2 - a2 * b1;
        float c1 = a2 * b0 - a0 * b2;
        float c2 = a0 * b1 - a1 * b0;
        float cn = sqrtf(c0 * c0 + c1 * c1 + c2 * c2);
        float inv2 = 1.0f / (1.0f + cn);
        float w2 = CWC[e * KCH + k];

        atomicAdd(&out[t * 384 + k],       r0 * inv1 * w1 + c0 * inv2 * w2);
        atomicAdd(&out[t * 384 + 128 + k], r1 * inv1 * w1 + c1 * inv2 * w2);
        atomicAdd(&out[t * 384 + 256 + k], r2 * inv1 * w1 + c2 * inv2 * w2);
    }
}

// ---------------------------------------------------------------------------
// Host-side orchestration
// ---------------------------------------------------------------------------
extern "C" void kernel_launch(void* const* d_in, const int* in_sizes, int n_in,
                              void* d_out, int out_size, void* d_ws, size_t ws_size,
                              hipStream_t stream) {
    const int*   batch   = (const int*)  d_in[0];
    const float* X       = (const float*)d_in[1];
    const float* H       = (const float*)d_in[2];
    const int*   eidx    = (const int*)  d_in[3];
    const float* eattr   = (const float*)d_in[4];
    const float* te      = (const float*)d_in[5];
    const float* W_hp    = (const float*)d_in[6];
    const float* b_hp    = (const float*)d_in[7];
    const float* gamma_h = (const float*)d_in[8];
    const float* beta_h  = (const float*)d_in[9];
    const float* w_xnorm = (const float*)d_in[10];
    const float* W_cp    = (const float*)d_in[11];
    const float* b_cp    = (const float*)d_in[12];
    const float* Wm1 = (const float*)d_in[13]; const float* bm1 = (const float*)d_in[14];
    const float* Wm2 = (const float*)d_in[15]; const float* bm2 = (const float*)d_in[16];
    const float* Wx1 = (const float*)d_in[17]; const float* bx1 = (const float*)d_in[18];
    const float* Wx2 = (const float*)d_in[19]; const float* bx2 = (const float*)d_in[20];
    const float* Wc1 = (const float*)d_in[21]; const float* bc1 = (const float*)d_in[22];
    const float* Wc2 = (const float*)d_in[23]; const float* bc2 = (const float*)d_in[24];

    const int N  = in_sizes[1] / 384;   // X is [N,3,128]
    const int E  = in_sizes[3] / 2;     // edge_index is [2,E]
    const int Bg = in_sizes[5] / 64;    // te is [B,64]
    float* out = (float*)d_out;
    float* ws  = (float*)d_ws;

    // workspace layout (floats; padded to 32B before Xn and before the f16 weights)
    size_t off = 0;
    float* sumX  = ws + off; off += (size_t)Bg * 384;
    float* cnt   = ws + off; off += (size_t)Bg;
    float* nsum  = ws + off; off += (size_t)Bg * KCH;
    float* sumXn = ws + off; off += (size_t)Bg * 384;   // becomes meanXn
    const int zeroCount = (int)off;
    off = (off + 7) & ~(size_t)7;
    float* Xn  = ws + off; off += (size_t)N * 384;
    float* Hn  = ws + off; off += (size_t)N * DE;
    float* dcv = ws + off; off += (size_t)N * 12;
    off = (off + 7) & ~(size_t)7;
    half_t* hb   = (half_t*)(ws + off);
    half_t* Wm1s = hb;                    // 288*64 (swizzled)
    half_t* Wm2s = Wm1s + MPAD * DE;      // 64*64
    half_t* Wx1s = Wm2s + DE * DE;
    half_t* Wx2s = Wx1s + DE * DE;        // 64*128
    half_t* Wc1s = Wx2s + DE * KCH;
    half_t* Wc2s = Wc1s + DE * DE;

    const int TB = 256;
    // 0) zero reduction scratch
    zero_kernel<<<(zeroCount + TB - 1) / TB, TB, 0, stream>>>(ws, zeroCount);
    // 1) weights -> f16, WMMA B-fragment order (Wm1 K-padded 260->288)
    swz_kernel<<<(MPAD * DE + TB - 1) / TB, TB, 0, stream>>>(Wm1, Wm1s, MIN_, MPAD, DE);
    swz_kernel<<<(DE * DE + TB - 1) / TB, TB, 0, stream>>>(Wm2, Wm2s, DE, DE, DE);
    swz_kernel<<<(DE * DE + TB - 1) / TB, TB, 0, stream>>>(Wx1, Wx1s, DE, DE, DE);
    swz_kernel<<<(DE * KCH + TB - 1) / TB, TB, 0, stream>>>(Wx2, Wx2s, DE, DE, KCH);
    swz_kernel<<<(DE * DE + TB - 1) / TB, TB, 0, stream>>>(Wc1, Wc1s, DE, DE, DE);
    swz_kernel<<<(DE * KCH + TB - 1) / TB, TB, 0, stream>>>(Wc2, Wc2s, DE, DE, KCH);
    // 2) segment reductions + E3Norm
    sumx_kernel<<<((N * 384) + TB - 1) / TB, TB, 0, stream>>>(X, batch, sumX, cnt, N);
    norm_kernel<<<((N * KCH) + TB - 1) / TB, TB, 0, stream>>>(X, batch, sumX, cnt, nsum, N);
    xn_kernel<<<((N * KCH) + TB - 1) / TB, TB, 0, stream>>>(X, batch, sumX, cnt, nsum,
                                                            w_xnorm, Xn, sumXn, out, N);
    meanxn_kernel<<<((Bg * 384) + TB - 1) / TB, TB, 0, stream>>>(sumXn, cnt, Bg);
    // 3) node features
    dc_kernel<<<((N * 12) + TB - 1) / TB, TB, 0, stream>>>(Xn, W_cp, b_cp, dcv, N);
    hproj_kernel<<<(N + 7) / 8, 256, 0, stream>>>(H, W_hp, b_hp, gamma_h, beta_h, Hn, N);
    // 4) WMMA edge MLPs + scatter
    edge_kernel<<<(E + TILE - 1) / TILE, 128, 0, stream>>>(
        eidx, eidx + E, batch, Hn, eattr, te, dcv, Xn, sumXn,
        Wm1s, bm1, Wm2s, bm2, Wx1s, bx1, Wx2s, bx2, Wc1s, bc1, Wc2s, bc2,
        out, E);
}
